// FastSpeech2_76922864271634
// MI455X (gfx1250) — matrix operand order repaired and checked by reading, not compiled, not run on hardware
//
#include <hip/hip_runtime.h>
#include <hip/hip_bf16.h>

typedef __attribute__((ext_vector_type(16))) _Float16 v16h;
typedef __attribute__((ext_vector_type(8)))  float    v8f;

#define BB    16
#define TT    256
#define DM    256
#define NH    4
#define DHD   64
#define FFD   1024
#define NMEL  80
#define MAXM  1024
#define MENC  (BB * TT)     /* 4096  */
#define MDEC  (BB * MAXM)   /* 16384 */

// ---------------------------------------------------------------------------
// Generic WMMA GEMM: out[M,N] = epilogue(A[M,K] @ Bw[K,N] + bias[N])
// MODE 0: plain   1: SiLU   2: ReLU   3: out = res + scale*(acc+bias)
// Block: 256 threads (8 waves), tile 64x64, K-step 32, f16 WMMA / f32 acc.
// ---------------------------------------------------------------------------
template<int MODE>
__global__ __launch_bounds__(256) void k_gemm(
    const float* __restrict__ A, const float* __restrict__ Bw,
    const float* __restrict__ bias, const float* res,
    float* out, int M, int N, int K, float scale)
{
    __shared__ _Float16 As[64][40];
    __shared__ _Float16 Bs[32][72];
    const int tid  = threadIdx.x;
    const int lane = tid & 31;
    const int wv   = tid >> 5;
    const int wn   = wv & 3;          // N sub-tile 0..3
    const int wm   = wv >> 2;         // M half 0..1
    const int nBlk = (N + 63) >> 6;
    const int bM   = (int)(blockIdx.x / nBlk) * 64;
    const int bN   = (int)(blockIdx.x % nBlk) * 64;
    const int ln15 = lane & 15;
    const int hi   = (lane >> 4) & 1;

    v8f acc0 = {}, acc1 = {};

    for (int k0 = 0; k0 < K; k0 += 32) {
#pragma unroll
        for (int i = 0; i < 8; ++i) {              // A tile 64x32
            int idx = tid + i * 256;
            int r = idx >> 5, c = idx & 31;
            int gr = bM + r;
            float v = (gr < M) ? A[(size_t)gr * K + (k0 + c)] : 0.0f;
            As[r][c] = (_Float16)v;
        }
#pragma unroll
        for (int i = 0; i < 8; ++i) {              // B tile 32x64
            int idx = tid + i * 256;
            int r = idx >> 6, c = idx & 63;
            int gc = bN + c;
            float v = (gc < N) ? Bw[(size_t)(k0 + r) * N + gc] : 0.0f;
            Bs[r][c] = (_Float16)v;
        }
        __syncthreads();
        v16h bf, a0, a1;
#pragma unroll
        for (int e = 0; e < 16; ++e)               // B frag: kk = hi*16 + e
            bf[e] = Bs[hi * 16 + e][wn * 16 + ln15];
#pragma unroll
        for (int e = 0; e < 16; ++e) {             // A frag: kk = hi*16 + e
            int kk = hi * 16 + e;
            a0[e] = As[wm * 32 + ln15][kk];
            a1[e] = As[wm * 32 + 16 + ln15][kk];
        }
        acc0 = __builtin_amdgcn_wmma_f32_16x16x32_f16(false, a0, false, bf, (short)0, acc0, false, false);
        acc1 = __builtin_amdgcn_wmma_f32_16x16x32_f16(false, a1, false, bf, (short)0, acc1, false, false);
        __syncthreads();
    }

    const int col = bN + wn * 16 + ln15;
    if (col < N) {
        const float bv = bias[col];
#pragma unroll
        for (int v = 0; v < 8; ++v) {
            int row0 = bM + wm * 32 + hi * 8 + v;
            int row1 = row0 + 16;
            if (row0 < M) {
                float x = acc0[v] + bv;
                if (MODE == 1) x = x / (1.0f + expf(-x));
                if (MODE == 2) x = fmaxf(x, 0.0f);
                if (MODE == 3) x = res[(size_t)row0 * N + col] + scale * x;
                out[(size_t)row0 * N + col] = x;
            }
            if (row1 < M) {
                float x = acc1[v] + bv;
                if (MODE == 1) x = x / (1.0f + expf(-x));
                if (MODE == 2) x = fmaxf(x, 0.0f);
                if (MODE == 3) x = res[(size_t)row1 * N + col] + scale * x;
                out[(size_t)row1 * N + col] = x;
            }
        }
    }
}

// ---------------------------------------------------------------------------
// Flash attention: one wave per (b, head, 16-query tile). dh=64, key tiles of
// 32. QK^T and PV via WMMA; online softmax with half-wave shfl reductions;
// P is re-laid-out C->A fragment through per-wave LDS.
// ---------------------------------------------------------------------------
__global__ __launch_bounds__(256) void k_flash(
    const float* __restrict__ Qb, const float* __restrict__ Kb,
    const float* __restrict__ Vb, float* __restrict__ Ob,
    const int* __restrict__ lens, int L, int nTask)
{
    __shared__ _Float16 Pls[8][16][40];
    const int w    = threadIdx.x >> 5;
    const int lane = threadIdx.x & 31;
    const int task = blockIdx.x * 8 + w;
    if (task >= nTask) return;                      // wave-uniform
    const int qtPerBH = L >> 4;
    const int qt = task % qtPerBH;
    const int bh = task / qtPerBH;
    const int h  = bh & (NH - 1);
    const int b  = bh / NH;
    const int seqLen = lens[b];
    const int ln15 = lane & 15;
    const int hi   = (lane >> 4) & 1;

    const float* qp = Qb + ((size_t)(b * L + qt * 16 + ln15)) * DM + h * DHD;
    v16h qa0, qa1;                                  // Q pre-scaled by 1/sqrt(64)
#pragma unroll
    for (int e = 0; e < 16; ++e) {
        int kk = hi * 16 + e;
        qa0[e] = (_Float16)(qp[kk] * 0.125f);
        qa1[e] = (_Float16)(qp[32 + kk] * 0.125f);
    }

    float mrun[8], lrun[8];
    v8f oacc[4];
    v8f zero = {};
#pragma unroll
    for (int v = 0; v < 8; ++v) { mrun[v] = -3.0e38f; lrun[v] = 0.0f; }
#pragma unroll
    for (int t = 0; t < 4; ++t) oacc[t] = zero;

    for (int kt = 0; kt < L; kt += 32) {
        v8f s0 = zero, s1 = zero;
        {
            const float* kp0 = Kb + ((size_t)(b * L + kt + ln15)) * DM + h * DHD + hi * 16;
            const float* kp1 = Kb + ((size_t)(b * L + kt + 16 + ln15)) * DM + h * DHD + hi * 16;
            v16h kb;
#pragma unroll
            for (int e = 0; e < 16; ++e) kb[e] = (_Float16)kp0[e];
            s0 = __builtin_amdgcn_wmma_f32_16x16x32_f16(false, qa0, false, kb, (short)0, s0, false, false);
#pragma unroll
            for (int e = 0; e < 16; ++e) kb[e] = (_Float16)kp0[32 + e];
            s0 = __builtin_amdgcn_wmma_f32_16x16x32_f16(false, qa1, false, kb, (short)0, s0, false, false);
#pragma unroll
            for (int e = 0; e < 16; ++e) kb[e] = (_Float16)kp1[e];
            s1 = __builtin_amdgcn_wmma_f32_16x16x32_f16(false, qa0, false, kb, (short)0, s1, false, false);
#pragma unroll
            for (int e = 0; e < 16; ++e) kb[e] = (_Float16)kp1[32 + e];
            s1 = __builtin_amdgcn_wmma_f32_16x16x32_f16(false, qa1, false, kb, (short)0, s1, false, false);
        }
        const bool msk0 = (kt + ln15) >= seqLen;
        const bool msk1 = (kt + 16 + ln15) >= seqLen;
        float p0[8], p1[8];
#pragma unroll
        for (int v = 0; v < 8; ++v) {
            float a = msk0 ? -1e9f : s0[v];
            float c = msk1 ? -1e9f : s1[v];
            s0[v] = a; s1[v] = c;
            float t = fmaxf(a, c);
#pragma unroll
            for (int o = 1; o < 16; o <<= 1) t = fmaxf(t, __shfl_xor(t, o, 32));
            float mnew = fmaxf(mrun[v], t);
            float alpha = expf(mrun[v] - mnew);
            mrun[v] = mnew;
            p0[v] = expf(s0[v] - mnew);
            p1[v] = expf(s1[v] - mnew);
            float rs = p0[v] + p1[v];
#pragma unroll
            for (int o = 1; o < 16; o <<= 1) rs += __shfl_xor(rs, o, 32);
            lrun[v] = alpha * lrun[v] + rs;
            oacc[0][v] *= alpha; oacc[1][v] *= alpha;
            oacc[2][v] *= alpha; oacc[3][v] *= alpha;
        }
        // C-layout P -> LDS -> A-fragment layout
#pragma unroll
        for (int v = 0; v < 8; ++v) {
            Pls[w][hi * 8 + v][ln15]      = (_Float16)p0[v];
            Pls[w][hi * 8 + v][16 + ln15] = (_Float16)p1[v];
        }
        v16h pa;
#pragma unroll
        for (int e = 0; e < 16; ++e) {
            int kk = hi * 16 + e;
            pa[e] = Pls[w][ln15][kk];
        }
        const float* vp = Vb + ((size_t)(b * L + kt + hi * 16)) * DM + h * DHD;
#pragma unroll
        for (int t4 = 0; t4 < 4; ++t4) {
            v16h vbf;
#pragma unroll
            for (int e = 0; e < 16; ++e)
                vbf[e] = (_Float16)vp[(size_t)e * DM + t4 * 16 + ln15];
            oacc[t4] = __builtin_amdgcn_wmma_f32_16x16x32_f16(false, pa, false, vbf, (short)0, oacc[t4], false, false);
        }
    }
#pragma unroll
    for (int v = 0; v < 8; ++v) {
        float inv = 1.0f / lrun[v];
        float* op = Ob + ((size_t)(b * L + qt * 16 + hi * 8 + v)) * DM + h * DHD + ln15;
#pragma unroll
        for (int t4 = 0; t4 < 4; ++t4) op[t4 * 16] = oacc[t4][v] * inv;
    }
}

// ---------------------------------------------------------------------------
// LayerNorm over D=256: one wave per row. MASKZ: zero rows >= lens[b].
// ACT 1: fused SiLU. Supports in-place (loads complete before stores).
// ---------------------------------------------------------------------------
template<int MASKZ, int ACT>
__global__ __launch_bounds__(256) void k_layernorm(
    const float* __restrict__ in, float* out,
    const float* __restrict__ g, const float* __restrict__ bb,
    const int* __restrict__ lens, int L, int M)
{
    int w = blockIdx.x * 8 + (threadIdx.x >> 5);
    int lane = threadIdx.x & 31;
    if (w >= M) return;
    const float* row = in + (size_t)w * DM;
    float x[8]; float s = 0.f;
#pragma unroll
    for (int i = 0; i < 8; ++i) { x[i] = row[i * 32 + lane]; s += x[i]; }
#pragma unroll
    for (int o = 1; o < 32; o <<= 1) s += __shfl_xor(s, o, 32);
    float mean = s * (1.0f / 256.0f);
    float vs = 0.f;
#pragma unroll
    for (int i = 0; i < 8; ++i) { float d = x[i] - mean; vs += d * d; }
#pragma unroll
    for (int o = 1; o < 32; o <<= 1) vs += __shfl_xor(vs, o, 32);
    float rstd = rsqrtf(vs * (1.0f / 256.0f) + 1e-5f);
    bool zero = false;
    if (MASKZ) zero = ((w % L) >= lens[w / L]);
    float* orow = out + (size_t)w * DM;
#pragma unroll
    for (int i = 0; i < 8; ++i) {
        int c = i * 32 + lane;
        float y = (x[i] - mean) * rstd * g[c] + bb[c];
        if (ACT == 1) y = y / (1.0f + expf(-y));
        orow[c] = zero ? 0.0f : y;
    }
}

// ------------------------- small elementwise kernels -----------------------
__global__ void k_embed(const int* __restrict__ ph, const float* __restrict__ emb, float* X)
{
    int i = blockIdx.x * 256 + threadIdx.x;
    if (i >= MENC * DM) return;
    int d = i & 255, tok = i >> 8;
    X[i] = emb[((size_t)ph[tok] << 8) + d];
}

__global__ void k_glu(const float* __restrict__ hin, float* out, int M)
{
    int i = blockIdx.x * 256 + threadIdx.x;
    if (i >= M * DM) return;
    int c = i & 255, m = i >> 8;
    float a = hin[(size_t)m * 512 + c];
    float gg = hin[(size_t)m * 512 + 256 + c];
    out[i] = a * (1.0f / (1.0f + expf(-gg)));
}

__global__ void k_dwconv(const float* __restrict__ gin, const float* __restrict__ wdw,
                         const float* __restrict__ bdw, float* out, int L, int M)
{
    int i = blockIdx.x * 256 + threadIdx.x;
    if (i >= M * DM) return;
    int c = i & 255, tok = i >> 8;
    int l = tok % L, b = tok / L;
    float acc = bdw[c];
#pragma unroll
    for (int t = 0; t < 31; ++t) {
        int ll = l + t - 15;
        if (ll >= 0 && ll < L)
            acc += gin[(((size_t)(b * L + ll)) << 8) + c] * wdw[t * 256 + c];
    }
    out[i] = acc;
}

__global__ void k_im2col3(const float* __restrict__ Xin, float* col, int nTok)
{
    long long i = (long long)blockIdx.x * 256 + threadIdx.x;
    if (i >= (long long)nTok * 768) return;
    int kd = (int)(i % 768), m = (int)(i / 768);
    int t = kd >> 8, ch = kd & 255;
    int l = m & 255, b = m >> 8;
    int ll = l + t - 1;
    float v = (ll >= 0 && ll < 256) ? Xin[((((size_t)(b << 8)) + ll) << 8) + ch] : 0.0f;
    col[(size_t)m * 768 + kd] = v;
}

__global__ __launch_bounds__(256) void k_rowdot(
    const float* __restrict__ Xin, const float* __restrict__ wv,
    const float* __restrict__ bv, float* out, int M)
{
    int w = blockIdx.x * 8 + (threadIdx.x >> 5);
    int lane = threadIdx.x & 31;
    if (w >= M) return;
    float s = 0.f;
#pragma unroll
    for (int i = 0; i < 8; ++i) {
        int c = i * 32 + lane;
        s += Xin[(size_t)w * DM + c] * wv[c];
    }
#pragma unroll
    for (int o = 1; o < 32; o <<= 1) s += __shfl_xor(s, o, 32);
    if (lane == 0) out[w] = s + bv[0];
}

__global__ void k_var_embed(float* X, const float* __restrict__ pv, const float* __restrict__ ev,
                            const float* __restrict__ wp, const float* __restrict__ bp,
                            const float* __restrict__ we, const float* __restrict__ be)
{
    int i = blockIdx.x * 256 + threadIdx.x;
    if (i >= MENC * DM) return;
    int d = i & 255, tok = i >> 8;
    int l = tok & 255, b = tok >> 8;
    float acc = X[i] + bp[d] + be[d];
#pragma unroll
    for (int t = 0; t < 9; ++t) {
        int ll = l + t - 4;
        if (ll >= 0 && ll < 256) {
            acc += pv[b * 256 + ll] * wp[t * 256 + d];
            acc += ev[b * 256 + ll] * we[t * 256 + d];
        }
    }
    X[i] = acc;
}

__global__ void k_cumdur(const int* __restrict__ dur, int* cum, int* mlen)
{
    int b = threadIdx.x;
    if (b >= BB) return;
    int s = 0;
    for (int t = 0; t < TT; ++t) {
        int d = dur[b * TT + t];
        if (d < 0) d = 0;
        s += d;
        cum[b * TT + t] = s;
    }
    mlen[b] = s < MAXM ? s : MAXM;
}

__global__ void k_regidx(const int* __restrict__ cum, int* ridx)
{
    int i = blockIdx.x * 256 + threadIdx.x;
    if (i >= BB * MAXM) return;
    int b = i >> 10, t = i & 1023;
    const int* c = cum + b * TT;
    int lo = 0, hiB = TT;
    while (lo < hiB) { int mid = (lo + hiB) >> 1; if (c[mid] <= t) lo = mid + 1; else hiB = mid; }
    ridx[i] = lo > TT - 1 ? TT - 1 : lo;
}

__global__ void k_gather(const float* __restrict__ Xe, const int* __restrict__ ridx,
                         const int* __restrict__ mlen, float* Xd)
{
    int i = blockIdx.x * 256 + threadIdx.x;
    if (i >= MDEC * DM) return;
    int d = i & 255, tok = i >> 8;
    int t = tok & 1023, b = tok >> 10;
    Xd[i] = (t < mlen[b]) ? Xe[(((size_t)(b * TT + ridx[tok])) << 8) + d] : 0.0f;
}

__global__ void k_melmask(const int* __restrict__ mlen, float* out)
{
    int i = blockIdx.x * 256 + threadIdx.x;
    if (i >= BB * MAXM) return;
    int b = i >> 10, t = i & 1023;
    out[i] = (t >= mlen[b]) ? 1.0f : 0.0f;
}

// ---------------------------------------------------------------------------
// Host orchestration.
// Param leaves assume dict-insertion-order flatten of setup_inputs():
//  0 phonemes  1 text_lengths  2 durations  3 embed
//  4.. encoder block0 (34 leaves), 38.. block1, 72.. dec block0, 106.. dec block1
//   per block: ff1{ln.g,ln.b,w1.w,w1.b,w2.w,w2.b} attn{ln.g,ln.b,q.w,q.b,k.w,k.b,
//   v.w,v.b,o.w,o.b} conv{ln.g,ln.b,pw1.w,pw1.b,dw.w,dw.b,ln2.g,ln2.b,pw2.w,pw2.b}
//   ff2{..6..} ln_out{g,b}
//  140 dur VP, 150 pitch VP, 160 energy VP (c1.w,c1.b,ln1.g,ln1.b,c2.w,c2.b,
//   ln2.g,ln2.b,proj.w,proj.b); 170/171 pitch_embed w/b; 172/173 energy_embed;
//  174/175 mel_proj w/b.
// ---------------------------------------------------------------------------
extern "C" void kernel_launch(void* const* d_in, const int* in_sizes, int n_in,
                              void* d_out, int out_size, void* d_ws, size_t ws_size,
                              hipStream_t stream)
{
    (void)in_sizes; (void)n_in; (void)out_size;
    auto F = [&](int i) { return (const float*)d_in[i]; };
    const int* ph   = (const int*)d_in[0];
    const int* tlen = (const int*)d_in[1];
    const int* dur  = (const int*)d_in[2];
    const float* emb = F(3);

    char* ws = (char*)d_ws;
    size_t off = 0;
    auto alloc = [&](size_t bytes) { void* p = ws + off; off += (bytes + 255) & ~(size_t)255; return p; };
    float* X  = (float*)alloc((size_t)MDEC * DM * 4);
    float* Hb = (float*)alloc((size_t)MDEC * DM * 4);
    float* T1 = (float*)alloc((size_t)MDEC * FFD * 4);
    float* Qb = (float*)alloc((size_t)MDEC * DM * 4);
    float* Kb = (float*)alloc((size_t)MDEC * DM * 4);
    float* Vb = (float*)alloc((size_t)MDEC * DM * 4);
    float* Ob = (float*)alloc((size_t)MDEC * DM * 4);
    float* Xd = (float*)alloc((size_t)MDEC * DM * 4);
    int* cum  = (int*)alloc(BB * TT * 4);
    int* mlen = (int*)alloc(BB * 4);
    int* ridx = (int*)alloc(BB * MAXM * 4);
    if (off > ws_size) return;

    float* outF   = (float*)d_out;
    float* melO   = outF;
    float* ldurO  = outF + (size_t)MDEC * NMEL;
    float* pitO   = ldurO + MENC;
    float* eneO   = pitO + MENC;
    float* mmaskO = eneO + MENC;

    auto gemm = [&](int mode, const float* Aa, const float* Bww, const float* bias,
                    const float* resp, float* outp, int M, int N, int K, float scale) {
        dim3 grid((unsigned)((M / 64) * ((N + 63) / 64)));
        switch (mode) {
        case 0: k_gemm<0><<<grid, 256, 0, stream>>>(Aa, Bww, bias, resp, outp, M, N, K, scale); break;
        case 1: k_gemm<1><<<grid, 256, 0, stream>>>(Aa, Bww, bias, resp, outp, M, N, K, scale); break;
        case 2: k_gemm<2><<<grid, 256, 0, stream>>>(Aa, Bww, bias, resp, outp, M, N, K, scale); break;
        default: k_gemm<3><<<grid, 256, 0, stream>>>(Aa, Bww, bias, resp, outp, M, N, K, scale); break;
        }
    };
    auto ln = [&](int mask, int act, const float* in, float* outp, const float* g,
                  const float* b, const int* lens, int L, int M) {
        dim3 grid((unsigned)((M + 7) / 8));
        if (mask)          k_layernorm<1, 0><<<grid, 256, 0, stream>>>(in, outp, g, b, lens, L, M);
        else if (act)      k_layernorm<0, 1><<<grid, 256, 0, stream>>>(in, outp, g, b, lens, L, M);
        else               k_layernorm<0, 0><<<grid, 256, 0, stream>>>(in, outp, g, b, lens, L, M);
    };
    auto g1 = [](int n) { return dim3((unsigned)((n + 255) / 256)); };

    auto run_block = [&](float* Xc, int pb, int M, int L, const int* lens) {
        // ff1 (macaron, 0.5 residual)
        ln(0, 0, Xc, Hb, F(pb + 0), F(pb + 1), nullptr, L, M);
        gemm(1, Hb, F(pb + 2), F(pb + 3), nullptr, T1, M, FFD, DM, 0.f);
        gemm(3, T1, F(pb + 4), F(pb + 5), Xc, Xc, M, DM, FFD, 0.5f);
        // MHSA
        ln(0, 0, Xc, Hb, F(pb + 6), F(pb + 7), nullptr, L, M);
        gemm(0, Hb, F(pb + 8),  F(pb + 9),  nullptr, Qb, M, DM, DM, 0.f);
        gemm(0, Hb, F(pb + 10), F(pb + 11), nullptr, Kb, M, DM, DM, 0.f);
        gemm(0, Hb, F(pb + 12), F(pb + 13), nullptr, Vb, M, DM, DM, 0.f);
        int nTask = M / 4;                          // (M/L)*NH*(L/16)
        k_flash<<<dim3((unsigned)((nTask + 7) / 8)), 256, 0, stream>>>(Qb, Kb, Vb, Ob, lens, L, nTask);
        gemm(3, Ob, F(pb + 14), F(pb + 15), Xc, Xc, M, DM, DM, 1.0f);
        // conv module
        ln(1, 0, Xc, Hb, F(pb + 16), F(pb + 17), lens, L, M);
        gemm(0, Hb, F(pb + 18), F(pb + 19), nullptr, T1, M, 2 * DM, DM, 0.f);
        k_glu<<<g1(M * DM), 256, 0, stream>>>(T1, Qb, M);
        k_dwconv<<<g1(M * DM), 256, 0, stream>>>(Qb, F(pb + 20), F(pb + 21), Kb, L, M);
        ln(0, 1, Kb, Vb, F(pb + 22), F(pb + 23), nullptr, L, M);
        gemm(3, Vb, F(pb + 24), F(pb + 25), Xc, Xc, M, DM, DM, 1.0f);
        // ff2
        ln(0, 0, Xc, Hb, F(pb + 26), F(pb + 27), nullptr, L, M);
        gemm(1, Hb, F(pb + 28), F(pb + 29), nullptr, T1, M, FFD, DM, 0.f);
        gemm(3, T1, F(pb + 30), F(pb + 31), Xc, Xc, M, DM, FFD, 0.5f);
        // ln_out + mask (in place)
        ln(1, 0, Xc, Xc, F(pb + 32), F(pb + 33), lens, L, M);
    };

    auto run_vp = [&](int vb, float* outp) {
        k_im2col3<<<g1(MENC * 3 * DM), 256, 0, stream>>>(X, T1, MENC);
        gemm(2, T1, F(vb + 0), F(vb + 1), nullptr, Qb, MENC, DM, 3 * DM, 0.f);
        ln(0, 0, Qb, Kb, F(vb + 2), F(vb + 3), nullptr, TT, MENC);
        k_im2col3<<<g1(MENC * 3 * DM), 256, 0, stream>>>(Kb, T1, MENC);
        gemm(2, T1, F(vb + 4), F(vb + 5), nullptr, Qb, MENC, DM, 3 * DM, 0.f);
        ln(0, 0, Qb, Kb, F(vb + 6), F(vb + 7), nullptr, TT, MENC);
        k_rowdot<<<dim3((unsigned)((MENC + 7) / 8)), 256, 0, stream>>>(Kb, F(vb + 8), F(vb + 9), outp, MENC);
    };

    // ---- encoder ----
    k_embed<<<g1(MENC * DM), 256, 0, stream>>>(ph, emb, X);
    run_block(X, 4,  MENC, TT, tlen);
    run_block(X, 38, MENC, TT, tlen);
    // ---- variance predictors ----
    run_vp(140, ldurO);
    run_vp(150, pitO);
    run_vp(160, eneO);
    k_var_embed<<<g1(MENC * DM), 256, 0, stream>>>(X, pitO, eneO, F(170), F(171), F(172), F(173));
    // ---- length regulation ----
    k_cumdur<<<dim3(1), 32, 0, stream>>>(dur, cum, mlen);
    k_regidx<<<g1(BB * MAXM), 256, 0, stream>>>(cum, ridx);
    k_gather<<<g1(MDEC * DM), 256, 0, stream>>>(X, ridx, mlen, Xd);
    k_melmask<<<g1(BB * MAXM), 256, 0, stream>>>(mlen, mmaskO);
    // ---- decoder ----
    run_block(Xd, 72,  MDEC, MAXM, mlen);
    run_block(Xd, 106, MDEC, MAXM, mlen);
    // ---- mel projection (N=80, guarded) ----
    gemm(0, Xd, F(174), F(175), nullptr, melO, MDEC, NMEL, DM, 0.f);
}
